// WaveNet_86122684220048
// MI455X (gfx1250) — matrix operand order repaired and checked by reading, not compiled, not run on hardware
//
#include <hip/hip_runtime.h>
#include <hip/hip_bf16.h>
#include <math.h>

// ---------------------------------------------------------------------------
// WaveNet on MI455X (gfx1250): all channel-mixing expressed as
// v_wmma_f32_16x16x32_f16. Time axis is parallel; layers are sequential
// kernel launches (h ping-pong). h kept in f16 (L2-resident), skip in f32.
// Transcendentals use raw v_exp_f32/v_rcp_f32 (branch-free, co-execute with
// WMMA since XDL matrix ops are tracked as TRANS ops on CDNA5).
// ---------------------------------------------------------------------------

typedef __attribute__((ext_vector_type(16))) _Float16 v16h;
typedef __attribute__((ext_vector_type(8)))  _Float16 v8h;
typedef __attribute__((ext_vector_type(8)))  float    v8f;

#define NSEQ 32768
#define NB   8
#define LAYM 19   // layers after the input layer

#define LOG2E 1.4426950408889634f
#define LN2   0.6931471805599453f

union V16U { v16h v; _Float16 h[16]; unsigned u[8]; };
union V8U  { v8h  v; _Float16 h[8];  unsigned u[4]; };

__device__ __forceinline__ v8f wmma_f16(v16h a, v16h b, v8f c) {
  return __builtin_amdgcn_wmma_f32_16x16x32_f16(
      /*neg_a=*/false, a, /*neg_b=*/false, b,
      /*c_mod=*/(short)0, c, /*reuse_a=*/false, /*reuse_b=*/false);
}

__device__ __forceinline__ unsigned pack2h(float a, float b) {
  union { _Float16 h[2]; unsigned u; } c;
  c.h[0] = (_Float16)a; c.h[1] = (_Float16)b;
  return c.u;
}

// Branch-free transcendentals -> v_exp_f32 / v_rcp_f32 / v_log_f32.
__device__ __forceinline__ float fast_exp(float x) {
  return __builtin_amdgcn_exp2f(x * LOG2E);
}
__device__ __forceinline__ float fast_sigmoid(float x) {
  // exp2(-inf) -> 0, rcp(inf) -> 0: saturates correctly without branches.
  return __builtin_amdgcn_rcpf(1.0f + __builtin_amdgcn_exp2f(-x * LOG2E));
}
__device__ __forceinline__ float fast_tanh(float x) {
  // tanh(x) = 1 - 2 / (exp(2x) + 1); saturates to +/-1 at extremes.
  float e = __builtin_amdgcn_exp2f(x * (2.0f * LOG2E));
  return __builtin_fmaf(-2.0f, __builtin_amdgcn_rcpf(e + 1.0f), 1.0f);
}

// ---------------------------------------------------------------------------
// Pack all A-operands (f16, exact per-lane WMMA A layout) into ws.
// Matrix ids: [0,19)=A_f  [19,38)=A_g  [38,57)=A_skip_lo  [57,76)=A_skip_hi
//             [76,80)=A_agg(m)  [80,112)=A_out(m,half)
// Each matrix: 32 lanes * 16 halves = 1 KB. Lane l<16 : M=l,  K(j)= j
//              Lane l>=16: M=l-16, K(j)= j+16   (contiguous K, matches B)
// ---------------------------------------------------------------------------
__global__ void wn_pack_kernel(const float* __restrict__ Wf,
                               const float* __restrict__ Wg,
                               const float* __restrict__ Wskip,
                               const float* __restrict__ Wagg,
                               const float* __restrict__ Wout,
                               _Float16* __restrict__ apack) {
  int m = blockIdx.x;          // 0..111
  int lane = threadIdx.x;      // 0..31
  int M = lane & 15;
  V16U out;
#pragma unroll
  for (int j = 0; j < 16; ++j) {
    int K = (lane < 16) ? j : j + 16;
    float v = 0.0f;
    if (m < 19) {                         // A_f: K<16 -> tap0 (t-d), K>=16 -> tap1 (t)
      int i = m;
      v = (K < 16) ? Wf[((i * 16 + M) * 16 + K) * 2 + 0]
                   : Wf[((i * 16 + M) * 16 + (K - 16)) * 2 + 1];
    } else if (m < 38) {
      int i = m - 19;
      v = (K < 16) ? Wg[((i * 16 + M) * 16 + K) * 2 + 0]
                   : Wg[((i * 16 + M) * 16 + (K - 16)) * 2 + 1];
    } else if (m < 57) {                  // skip lo: only current-time half
      int i = m - 38;
      v = (K >= 16) ? Wskip[(i * 32 + M) * 16 + (K - 16)] : 0.0f;
    } else if (m < 76) {                  // skip hi
      int i = m - 57;
      v = (K >= 16) ? Wskip[(i * 32 + 16 + M) * 16 + (K - 16)] : 0.0f;
    } else if (m < 80) {                  // agg tile m: 16 out ch x 32 skip ch
      int mm = m - 76;
      v = Wagg[(mm * 16 + M) * 32 + K];
    } else {                              // out tile (mt, half): K over 64 agg ch
      int q = m - 80, mt = q >> 1, half = q & 1;
      v = Wout[(mt * 16 + M) * 64 + half * 32 + K];
    }
    out.h[j] = (_Float16)v;
  }
  *(v16h*)(apack + ((size_t)m * 32 + lane) * 16) = out.v;
}

// ---------------------------------------------------------------------------
// Input layer: h0[b][t][c] = xs[t] + W0[c,0]*xs[t-1] + W0[c,1]*xs[t] + b0[c]
// with xs[t] = (t>0 ? x[t-1] : 0) / 32768.  Time-major f16 so WMMA B-operand
// slices are contiguous 32-byte rows.
// ---------------------------------------------------------------------------
__global__ void wn_init_kernel(const int* __restrict__ xint,
                               const float* __restrict__ W0,
                               const float* __restrict__ b0,
                               _Float16* __restrict__ h) {
  int idx = blockIdx.x * blockDim.x + threadIdx.x;
  if (idx >= NB * NSEQ) return;
  int t = idx & (NSEQ - 1);
  int b = idx >> 15;
  const int* xr = xint + (size_t)b * NSEQ;
  float xs  = (t >= 1) ? (float)xr[t - 1] * (1.0f / 32768.0f) : 0.0f;  // xs[t]
  float xs1 = (t >= 2) ? (float)xr[t - 2] * (1.0f / 32768.0f) : 0.0f;  // xs[t-1]
  V16U o;
#pragma unroll
  for (int c = 0; c < 16; ++c) {
    float v = xs + W0[c * 2 + 0] * xs1 + W0[c * 2 + 1] * xs + b0[c];
    o.h[c] = (_Float16)v;
  }
  *(v16h*)(h + (size_t)idx * 16) = o.v;
}

// ---------------------------------------------------------------------------
// One residual layer. One wave = one (batch, 16-timestep) tile.
//   B (32x16): lane l<16 -> h(t0+l-d, ch0..15); lane l+16 -> h(t0+l, ch0..15)
//   4 WMMAs: f, g, skip_lo, skip_hi.  h_new = h + tanh(f)*sigmoid(g).
// ---------------------------------------------------------------------------
__global__ __launch_bounds__(256)
void wn_layer_kernel(const _Float16* __restrict__ hin,
                     _Float16* __restrict__ hout,
                     float* __restrict__ skip,
                     const _Float16* __restrict__ apack,
                     const float* __restrict__ bf,
                     const float* __restrict__ bg,
                     const float* __restrict__ bsk,
                     int layer, int d, int accumulate) {
  int gtid = blockIdx.x * blockDim.x + threadIdx.x;
  int wave = gtid >> 5;
  int lane = threadIdx.x & 31;
  int b  = wave >> 11;                 // 2048 tiles per batch
  int t0 = (wave & 2047) << 4;
  bool hi = lane >= 16;
  int tl = t0 + (lane & 15);

  const v16h* A = (const v16h*)apack;
  v16h a_f  = A[(size_t)(0  + layer) * 32 + lane];
  v16h a_g  = A[(size_t)(19 + layer) * 32 + lane];
  v16h a_sl = A[(size_t)(38 + layer) * 32 + lane];
  v16h a_sh = A[(size_t)(57 + layer) * 32 + lane];

  // B operand: contiguous 16-channel f16 row per lane.
  const v16h* hrow = (const v16h*)(hin + (size_t)b * NSEQ * 16);
  v16h bmat = {};
  if (!hi) {
    int tp = tl - d;
    if (tp >= 0) bmat = hrow[tp];
  } else {
    bmat = hrow[tl];
  }

  v8f zero = {};
  v8f df  = wmma_f16(a_f,  bmat, zero);
  v8f dg  = wmma_f16(a_g,  bmat, zero);
  v8f dsl = wmma_f16(a_sl, bmat, zero);
  v8f dsh = wmma_f16(a_sh, bmat, zero);

  int c0 = hi ? 8 : 0;  // D reg r holds channel c0+r at time tl
  V8U hold = *(const V8U*)(hin + ((size_t)b * NSEQ + tl) * 16 + c0);
  V8U hnew;
#pragma unroll
  for (int r = 0; r < 8; ++r) {
    float f = fast_tanh(df[r] + bf[layer * 16 + c0 + r]);
    float g = fast_sigmoid(dg[r] + bg[layer * 16 + c0 + r]);
    hnew.h[r] = (_Float16)((float)hold.h[r] + f * g);
  }
  *(v8h*)(hout + ((size_t)b * NSEQ + tl) * 16 + c0) = hnew.v;

  // skip accumulation (f32, time-major rows of 32)
  float* srow = skip + ((size_t)b * NSEQ + tl) * 32;
#pragma unroll
  for (int r = 0; r < 8; ++r) {
    float vlo = dsl[r] + bsk[layer * 32 + c0 + r];
    float vhi = dsh[r] + bsk[layer * 32 + 16 + c0 + r];
    if (accumulate) {
      vlo += srow[c0 + r];
      vhi += srow[16 + c0 + r];
    }
    srow[c0 + r] = vlo;
    srow[16 + c0 + r] = vhi;
  }
}

// ---------------------------------------------------------------------------
// Head: relu(skip) -> agg(64x32) -> relu -> out(256x64) -> log_softmax.
// agg D-layout -> B-layout needs only a lane<->lane+16 swap of packed f16.
// ---------------------------------------------------------------------------
__global__ __launch_bounds__(128)
void wn_head_kernel(const float* __restrict__ skip,
                    const _Float16* __restrict__ apack,
                    const float* __restrict__ bagg,
                    const float* __restrict__ bout,
                    float* __restrict__ out) {
  int gtid = blockIdx.x * blockDim.x + threadIdx.x;
  int wave = gtid >> 5;
  int lane = threadIdx.x & 31;
  int b  = wave >> 11;
  int t0 = (wave & 2047) << 4;
  bool hi = lane >= 16;
  int tl = t0 + (lane & 15);
  int c0 = hi ? 8 : 0;

  // B_skip: lane l<16 -> relu(skip[tl][0..15]); lane l+16 -> relu(skip[tl][16..31])
  const float* srow = skip + ((size_t)b * NSEQ + tl) * 32 + (hi ? 16 : 0);
  V16U bsk;
#pragma unroll
  for (int j = 0; j < 16; ++j) {
    float v = srow[j];
    bsk.h[j] = (_Float16)(v > 0.0f ? v : 0.0f);
  }

  const v16h* A = (const v16h*)apack;
  v8f zero = {};
  unsigned pk[4][4];
#pragma unroll
  for (int m = 0; m < 4; ++m) {
    v16h a = A[(size_t)(76 + m) * 32 + lane];
    v8f dag = wmma_f16(a, bsk.v, zero);
    // bias + relu, pack to f16 pairs for the lane swap
#pragma unroll
    for (int p = 0; p < 4; ++p) {
      float v0 = dag[2 * p + 0] + bagg[m * 16 + c0 + 2 * p + 0];
      float v1 = dag[2 * p + 1] + bagg[m * 16 + c0 + 2 * p + 1];
      v0 = v0 > 0.0f ? v0 : 0.0f;
      v1 = v1 > 0.0f ? v1 : 0.0f;
      pk[m][p] = pack2h(v0, v1);
    }
  }
  unsigned xo[4][4];
#pragma unroll
  for (int m = 0; m < 4; ++m)
#pragma unroll
    for (int p = 0; p < 4; ++p)
      xo[m][p] = (unsigned)__shfl_xor((int)pk[m][p], 16, 32);

  // B1 = agg ch 0..31, B2 = agg ch 32..63 (K over agg channels)
  V16U b1, b2;
#pragma unroll
  for (int p = 0; p < 4; ++p) {
    b1.u[p]     = hi ? xo[1][p] : pk[0][p];
    b1.u[4 + p] = hi ? pk[1][p] : xo[0][p];
    b2.u[p]     = hi ? xo[3][p] : pk[2][p];
    b2.u[4 + p] = hi ? pk[3][p] : xo[2][p];
  }

  // 256 output classes: 16 M-tiles, K=64 via two chained WMMAs each
  v8f lg[16];
#pragma unroll
  for (int m = 0; m < 16; ++m) {
    v16h alo = A[(size_t)(80 + 2 * m + 0) * 32 + lane];
    v16h ahi = A[(size_t)(80 + 2 * m + 1) * 32 + lane];
    v8f acc = wmma_f16(alo, b1.v, zero);
    acc = wmma_f16(ahi, b2.v, acc);
#pragma unroll
    for (int r = 0; r < 8; ++r)
      acc[r] = acc[r] + bout[m * 16 + c0 + r];
    lg[m] = acc;
  }

  // log_softmax over 256 classes: 128 values here, 128 in lane^16
  float mx = -3.4e38f;
#pragma unroll
  for (int m = 0; m < 16; ++m)
#pragma unroll
    for (int r = 0; r < 8; ++r) mx = fmaxf(mx, lg[m][r]);
  mx = fmaxf(mx, __shfl_xor(mx, 16, 32));
  float s = 0.0f;
#pragma unroll
  for (int m = 0; m < 16; ++m)
#pragma unroll
    for (int r = 0; r < 8; ++r)
      s += __builtin_amdgcn_exp2f((lg[m][r] - mx) * LOG2E);  // arg <= 0, safe
  s += __shfl_xor(s, 16, 32);
  float lz = __builtin_amdgcn_logf(s) * LN2;

  float* orow = out + (size_t)b * 256 * NSEQ + tl;
#pragma unroll
  for (int m = 0; m < 16; ++m)
#pragma unroll
    for (int r = 0; r < 8; ++r)
      orow[(size_t)(m * 16 + c0 + r) * NSEQ] = lg[m][r] - mx - lz;
}

// ---------------------------------------------------------------------------
extern "C" void kernel_launch(void* const* d_in, const int* in_sizes, int n_in,
                              void* d_out, int out_size, void* d_ws, size_t ws_size,
                              hipStream_t stream) {
  const int*   xint  = (const int*)d_in[0];
  const float* W0    = (const float*)d_in[1];
  const float* b0    = (const float*)d_in[2];
  const float* Wf    = (const float*)d_in[3];
  const float* bf    = (const float*)d_in[4];
  const float* Wg    = (const float*)d_in[5];
  const float* bg    = (const float*)d_in[6];
  const float* Wskip = (const float*)d_in[7];
  const float* bsk   = (const float*)d_in[8];
  const float* Wagg  = (const float*)d_in[9];
  const float* bagg  = (const float*)d_in[10];
  const float* Wout  = (const float*)d_in[11];
  const float* bout  = (const float*)d_in[12];
  (void)in_sizes; (void)n_in; (void)out_size; (void)ws_size;

  char* ws = (char*)d_ws;
  _Float16* apack = (_Float16*)ws;                               // 112 KB
  _Float16* h0    = (_Float16*)(ws + (size_t)(128 << 10));       // 8 MB
  _Float16* h1    = (_Float16*)(ws + (size_t)(128 << 10) + ((size_t)8 << 20));
  float*    skip  = (float*)   (ws + (size_t)(128 << 10) + ((size_t)16 << 20)); // 32 MB

  wn_pack_kernel<<<112, 32, 0, stream>>>(Wf, Wg, Wskip, Wagg, Wout, apack);
  wn_init_kernel<<<(NB * NSEQ + 255) / 256, 256, 0, stream>>>(xint, W0, b0, h0);

  _Float16* hin = h0;
  _Float16* hout = h1;
  for (int i = 0; i < LAYM; ++i) {
    int d = 1 << ((i + 1) % 10);
    wn_layer_kernel<<<2048, 256, 0, stream>>>(hin, hout, skip, apack,
                                              bf, bg, bsk, i, d, (i > 0) ? 1 : 0);
    _Float16* tmp = hin; hin = hout; hout = tmp;
  }

  wn_head_kernel<<<4096, 128, 0, stream>>>(skip, apack, bagg, bout, (float*)d_out);
}
